// YoutubeSBC_36069135352387
// MI455X (gfx1250) — compile-verified, matrix operand order checked by reading, not run on hardware
//
#include <hip/hip_runtime.h>

typedef __attribute__((ext_vector_type(16))) _Float16 v16h;
typedef __attribute__((ext_vector_type(8)))  _Float16 v8h;
typedef __attribute__((ext_vector_type(8)))  float    v8f;
typedef __attribute__((ext_vector_type(4)))  float    v4f;

#define B_ROWS 4096
#define NFEAT  4
#define VOCAB  100000
#define EMB    64
#define D_IN   256   // NFEAT*EMB
#define D_H    256
#define D_O    128

// ---------------------------------------------------------------------------
// Weight prep: Wt[n][k] = (f16) W[k][n]   (so B-fragments are K-contiguous)
// ---------------------------------------------------------------------------
__global__ void wt_kernel(const float* __restrict__ W, _Float16* __restrict__ Wt,
                          int K, int N) {
    int tid = blockIdx.x * blockDim.x + threadIdx.x;
    if (tid >= K * N) return;
    int k = tid / N, n = tid - k * N;
    Wt[(size_t)n * K + k] = (_Float16)W[(size_t)k * N + n];
}

// ---------------------------------------------------------------------------
// Embedding gather for both towers -> f16 [4096][256]
// ---------------------------------------------------------------------------
__global__ void embed_kernel(const int* __restrict__ uids, const int* __restrict__ iids,
                             const float* __restrict__ utab, const float* __restrict__ itab,
                             _Float16* __restrict__ xu, _Float16* __restrict__ xi) {
    int t = blockIdx.x * blockDim.x + threadIdx.x;
    int tower = (t >= B_ROWS * D_IN) ? 1 : 0;
    int idx = tower ? (t - B_ROWS * D_IN) : t;
    int row = idx >> 8;          // /256
    int within = idx & 255;
    int f = within >> 6;         // /64
    int e = within & 63;
    const int*   ids = tower ? iids : uids;
    const float* tab = tower ? itab : utab;
    _Float16*    x   = tower ? xi : xu;
    int id = ids[row * NFEAT + f];
    float val = tab[((size_t)f * VOCAB + (size_t)id) * EMB + e];
    x[(size_t)row * D_IN + within] = (_Float16)val;
}

// ---------------------------------------------------------------------------
// Fused two-layer MLP tower + row L2-normalization.
// Block = 128 threads (4 waves), handles 16 rows.
//   h  = relu(x  @ W1 + b1)   16x256, f16 staged in LDS
//   u  = relu(h  @ W2 + b2)   16x128, f32 in LDS
//   un = u / max(||u||, 1e-8) -> global f32
// WMMA fragment layout (16-bit A, wave32):
//   lanes 0-15 : row=lane,    halves [K0..7 | K16..23]
//   lanes 16-31: row=lane-16, halves [K8..15| K24..31]
// B is stored n-major [N][K] so its fragment uses the identical pattern.
// ---------------------------------------------------------------------------
__launch_bounds__(128)
__global__ void mlp_kernel(const _Float16* __restrict__ x,    // [4096][256]
                           const _Float16* __restrict__ W1t,  // [256][256] n-major
                           const float*    __restrict__ b1,   // [256]
                           const _Float16* __restrict__ W2t,  // [128][256] n-major
                           const float*    __restrict__ b2,   // [128]
                           float* __restrict__ out_n)         // [4096][128]
{
    __shared__ _Float16 x_lds[16][D_IN];
    __shared__ _Float16 h_lds[16][D_H];
    __shared__ float    u_lds[16][D_O];
    __shared__ float    rnorm[16];

    const int tid  = threadIdx.x;
    const int wave = tid >> 5;
    const int lane = tid & 31;
    const int half = lane >> 4;   // 0: lanes 0-15, 1: lanes 16-31
    const int idx  = lane & 15;
    const int row0 = blockIdx.x * 16;

    // Stage x tile (16x256 f16 = 512 x b128 chunks)
    {
        const v8h* src = reinterpret_cast<const v8h*>(x + (size_t)row0 * D_IN);
        v8h* dst = reinterpret_cast<v8h*>(&x_lds[0][0]);
        for (int i = tid; i < 16 * D_IN / 8; i += 128) dst[i] = src[i];
    }
    __syncthreads();

    // ---- GEMM1: 16 n-tiles, 4 per wave, K=256 chained WMMA ----
    for (int nt = wave; nt < D_H / 16; nt += 4) {
        const int n = nt * 16 + idx;
        v8f c = {};
        for (int k0 = 0; k0 < D_IN; k0 += 32) {
            v8h alo = *reinterpret_cast<const v8h*>(&x_lds[idx][k0 + half * 8]);
            v8h ahi = *reinterpret_cast<const v8h*>(&x_lds[idx][k0 + 16 + half * 8]);
            v8h blo = *reinterpret_cast<const v8h*>(W1t + (size_t)n * D_IN + k0 + half * 8);
            v8h bhi = *reinterpret_cast<const v8h*>(W1t + (size_t)n * D_IN + k0 + 16 + half * 8);
            v16h a, b;
            for (int i = 0; i < 8; ++i) {
                a[i] = alo[i]; a[8 + i] = ahi[i];
                b[i] = blo[i]; b[8 + i] = bhi[i];
            }
            c = __builtin_amdgcn_wmma_f32_16x16x32_f16(false, a, false, b,
                                                       (short)0, c, false, false);
        }
        const float bn = b1[n];
        for (int r = 0; r < 8; ++r) {
            float v = c[r] + bn;
            h_lds[half * 8 + r][n] = (_Float16)(v > 0.f ? v : 0.f);
        }
    }
    __syncthreads();

    // ---- GEMM2: 8 n-tiles, 2 per wave ----
    for (int nt = wave; nt < D_O / 16; nt += 4) {
        const int n = nt * 16 + idx;
        v8f c = {};
        for (int k0 = 0; k0 < D_H; k0 += 32) {
            v8h alo = *reinterpret_cast<const v8h*>(&h_lds[idx][k0 + half * 8]);
            v8h ahi = *reinterpret_cast<const v8h*>(&h_lds[idx][k0 + 16 + half * 8]);
            v8h blo = *reinterpret_cast<const v8h*>(W2t + (size_t)n * D_H + k0 + half * 8);
            v8h bhi = *reinterpret_cast<const v8h*>(W2t + (size_t)n * D_H + k0 + 16 + half * 8);
            v16h a, b;
            for (int i = 0; i < 8; ++i) {
                a[i] = alo[i]; a[8 + i] = ahi[i];
                b[i] = blo[i]; b[8 + i] = bhi[i];
            }
            c = __builtin_amdgcn_wmma_f32_16x16x32_f16(false, a, false, b,
                                                       (short)0, c, false, false);
        }
        const float bn = b2[n];
        for (int r = 0; r < 8; ++r) {
            float v = c[r] + bn;
            u_lds[half * 8 + r][n] = (v > 0.f ? v : 0.f);
        }
    }
    __syncthreads();

    // ---- Row L2 norms (16 rows handled by 16 lanes of wave 0) ----
    if (tid < 16) {
        float s = 0.f;
        for (int k = 0; k < D_O; ++k) { float v = u_lds[tid][k]; s += v * v; }
        float nrm = sqrtf(s);
        rnorm[tid] = (nrm > 1e-8f) ? nrm : 1e-8f;
    }
    __syncthreads();

    for (int i = tid; i < 16 * D_O; i += 128) {
        int r = i >> 7, cc = i & (D_O - 1);
        out_n[(size_t)(row0 + r) * D_O + cc] = u_lds[r][cc] / rnorm[r];
    }
}

// ---------------------------------------------------------------------------
// Only 4 shifted columns of the 4096x4096 cosine matrix are needed:
// out[i][j] = dot(un[i], vn[(i+j)%B]) - log(sw[(i+j)%B])
// One wave per row; 128-dim dot spread 4 floats/lane, shuffle reduction.
// ---------------------------------------------------------------------------
__global__ void score_kernel(const float* __restrict__ un, const float* __restrict__ vn,
                             const int* __restrict__ sw_ids,
                             const float* __restrict__ sw_table,
                             float* __restrict__ out)
{
    const int wave = threadIdx.x >> 5;
    const int lane = threadIdx.x & 31;
    const int row  = blockIdx.x * 8 + wave;
    v4f a = *reinterpret_cast<const v4f*>(un + (size_t)row * D_O + lane * 4);
    for (int j = 0; j < 4; ++j) {
        const int col = (row + j) & (B_ROWS - 1);
        v4f b = *reinterpret_cast<const v4f*>(vn + (size_t)col * D_O + lane * 4);
        float s = a[0] * b[0] + a[1] * b[1] + a[2] * b[2] + a[3] * b[3];
        for (int m = 16; m >= 1; m >>= 1) s += __shfl_xor(s, m, 32);
        if (lane == 0) {
            out[(size_t)row * 4 + j] = s - logf(sw_table[sw_ids[col]]);
        }
    }
}

// ---------------------------------------------------------------------------
extern "C" void kernel_launch(void* const* d_in, const int* in_sizes, int n_in,
                              void* d_out, int out_size, void* d_ws, size_t ws_size,
                              hipStream_t stream)
{
    const int*   user_ids = (const int*)  d_in[0];
    const int*   item_ids = (const int*)  d_in[1];
    const int*   sw_ids   = (const int*)  d_in[2];
    const float* utab     = (const float*)d_in[3];
    const float* itab     = (const float*)d_in[4];
    const float* sw_table = (const float*)d_in[5];
    const float* uW1 = (const float*)d_in[6];
    const float* ub1 = (const float*)d_in[7];
    const float* uW2 = (const float*)d_in[8];
    const float* ub2 = (const float*)d_in[9];
    const float* iW1 = (const float*)d_in[10];
    const float* ib1 = (const float*)d_in[11];
    const float* iW2 = (const float*)d_in[12];
    const float* ib2 = (const float*)d_in[13];

    char* ws = (char*)d_ws;
    size_t off = 0;
    auto alloc = [&](size_t bytes) -> void* {
        void* p = ws + off;
        off = (off + bytes + 255) & ~(size_t)255;
        return p;
    };
    _Float16* xu   = (_Float16*)alloc((size_t)B_ROWS * D_IN * 2);
    _Float16* xi   = (_Float16*)alloc((size_t)B_ROWS * D_IN * 2);
    _Float16* W1tu = (_Float16*)alloc((size_t)D_H * D_IN * 2);
    _Float16* W2tu = (_Float16*)alloc((size_t)D_O * D_H * 2);
    _Float16* W1ti = (_Float16*)alloc((size_t)D_H * D_IN * 2);
    _Float16* W2ti = (_Float16*)alloc((size_t)D_O * D_H * 2);
    float*    un   = (float*)   alloc((size_t)B_ROWS * D_O * 4);
    float*    vn   = (float*)   alloc((size_t)B_ROWS * D_O * 4);

    // Weight transpose+convert (tiny)
    wt_kernel<<<(D_IN * D_H + 255) / 256, 256, 0, stream>>>(uW1, W1tu, D_IN, D_H);
    wt_kernel<<<(D_H * D_O + 255) / 256, 256, 0, stream>>>(uW2, W2tu, D_H, D_O);
    wt_kernel<<<(D_IN * D_H + 255) / 256, 256, 0, stream>>>(iW1, W1ti, D_IN, D_H);
    wt_kernel<<<(D_H * D_O + 255) / 256, 256, 0, stream>>>(iW2, W2ti, D_H, D_O);

    // Embedding gather, both towers
    embed_kernel<<<(2 * B_ROWS * D_IN) / 256, 256, 0, stream>>>(
        user_ids, item_ids, utab, itab, xu, xi);

    // Fused MLP + normalize per tower (WMMA path)
    mlp_kernel<<<B_ROWS / 16, 128, 0, stream>>>(xu, W1tu, ub1, W2tu, ub2, un);
    mlp_kernel<<<B_ROWS / 16, 128, 0, stream>>>(xi, W1ti, ib1, W2ti, ib2, vn);

    // Sampled-softmax scoring (only the 4 needed columns per row)
    score_kernel<<<B_ROWS / 8, 256, 0, stream>>>(un, vn, sw_ids, sw_table, (float*)d_out);
}
